// CA_Module_81492709474559
// MI455X (gfx1250) — matrix operand order repaired and checked by reading, not compiled, not run on hardware
//
#include <hip/hip_runtime.h>
#include <hip/hip_bf16.h>

typedef __attribute__((ext_vector_type(16))) __bf16 v16bf;
typedef __attribute__((ext_vector_type(8)))  __bf16 v8bf;
typedef __attribute__((ext_vector_type(4)))  __bf16 v4bf;
typedef __attribute__((ext_vector_type(8)))  float  v8f;
typedef __attribute__((ext_vector_type(8)))  unsigned short u16x8;

constexpr int Bn  = 16;
constexpr int Cn  = 512;
constexpr int HWN = 4096;
constexpr int KC  = 32;   // K elems staged per buffer
constexpr int LP  = 40;   // padded LDS row stride (bf16 elems), keeps 16B alignment

typedef const __attribute__((address_space(1))) char* gas1_t;
typedef __attribute__((address_space(3))) char* las3_t;

#if __has_builtin(__builtin_amdgcn_global_load_async_to_lds_b128) && \
    __has_builtin(__builtin_amdgcn_s_wait_asynccnt)
#define HAS_ASYNC_LDS 1
typedef int vi4 __attribute__((vector_size(16)));
typedef __attribute__((address_space(1))) vi4* gvi4_t;
typedef __attribute__((address_space(3))) vi4* lvi4_t;
#else
#define HAS_ASYNC_LDS 0
#endif

#if __has_builtin(__builtin_amdgcn_tensor_load_to_lds) && \
    __has_builtin(__builtin_amdgcn_s_wait_tensorcnt)
#define HAS_TDM 1
typedef unsigned int u32x4 __attribute__((vector_size(16)));
typedef int i32x4 __attribute__((vector_size(16)));
typedef int i32x8 __attribute__((vector_size(32)));
#else
#define HAS_TDM 0
#endif

__device__ __forceinline__ void wait_stage() {
#if HAS_ASYNC_LDS
  __builtin_amdgcn_s_wait_asynccnt(0);
#endif
}

#if HAS_TDM
// TDM: DMA a [128 rows x 32 bf16] tile (global row stride = gstride elems) into
// LDS at s with the LP=40 padded layout (pad 4 dwords after every 16 dwords).
// Issue from ONE wave only; complete with s_wait_tensorcnt + barrier.
__device__ __forceinline__ void tdm_stage_rows(const __bf16* g, unsigned gstride,
                                               __bf16* s) {
  const unsigned long long ga = (unsigned long long)(const void*)g;
  const unsigned lds = (unsigned)(unsigned long long)(las3_t)(char*)s;
  u32x4 g0;
  g0[0] = 1u;                                   // count=1, user descriptor
  g0[1] = lds;                                  // lds_addr (bytes)
  g0[2] = (unsigned)ga;                         // global_addr[31:0]
  g0[3] = (unsigned)(ga >> 32) | (2u << 30);    // global_addr[56:32] | type=2
  const unsigned dim0 = gstride;                // tensor dim0 (elems)
  const unsigned dim1 = 512;                    // tensor dim1 (rows)
  i32x8 g1;
  g1[0] = (1 << 16)                             // data_size = 2 bytes
        | (1 << 20)                             // pad_enable
        | (3 << 22)                             // pad_interval: 16 dwords (64B)
        | (3 << 25);                            // pad_amount:   4 dwords (16B)
  g1[1] = (int)((dim0 & 0xFFFFu) << 16);        // tensor_dim0[15:0]
  g1[2] = (int)((dim0 >> 16) & 0xFFFFu) | (int)((dim1 & 0xFFFFu) << 16);
  g1[3] = (int)((dim1 >> 16) & 0xFFFFu) | (KC << 16);   // tile_dim0 = 32
  g1[4] = 128;                                  // tile_dim1 = 128 rows
  g1[5] = (int)gstride;                         // tensor_dim0_stride lo32
  g1[6] = 0;                                    // stride hi16 | dim1_stride lo16
  g1[7] = 0;
  i32x4 z4 = {0, 0, 0, 0};
#if __clang_major__ >= 23
  i32x8 z8 = {0, 0, 0, 0, 0, 0, 0, 0};
  __builtin_amdgcn_tensor_load_to_lds(g0, g1, z4, z4, z8, 0);
#else
  __builtin_amdgcn_tensor_load_to_lds(g0, g1, z4, z4, 0);
#endif
}
#endif

// Stage 128 rows x 32 bf16 (row-major, global row stride = gstride elems) into
// s[row*LP + k]. 256 threads: 2 threads per row, 32B each.
__device__ __forceinline__ void stage_rows(const __bf16* g, int gstride,
                                           __bf16* s, int t) {
  const int r = t >> 1, seg = (t & 1) * 16;
  const __bf16* src = g + (size_t)r * gstride + seg;
  __bf16* dst = s + r * LP + seg;
#if HAS_ASYNC_LDS
  gvi4_t gp = (gvi4_t)(gas1_t)(const char*)src;
  lvi4_t sp = (lvi4_t)(las3_t)(char*)dst;
  __builtin_amdgcn_global_load_async_to_lds_b128(gp, sp, 0, 0);
  __builtin_amdgcn_global_load_async_to_lds_b128(gp + 1, sp + 1, 0, 0);
#else
  ((uint4*)dst)[0] = ((const uint4*)src)[0];
  ((uint4*)dst)[1] = ((const uint4*)src)[1];
#endif
}

// Stage a 32(k) x 128(n) tile of Q (row stride HWN) transposed into
// s[n*LP + k]. Packs two k's per 32-bit DS store.
__device__ __forceinline__ void stage_transpose(const __bf16* g, __bf16* s, int t) {
  const int k2  = (t >> 4) * 2;     // 0..30
  const int nch = (t & 15) * 8;     // 0..120
  const u16x8 r0 = *(const u16x8*)(g + (size_t)k2 * HWN + nch);
  const u16x8 r1 = *(const u16x8*)(g + (size_t)(k2 + 1) * HWN + nch);
  for (int j = 0; j < 8; ++j) {
    const unsigned pk = (unsigned)r0[j] | ((unsigned)r1[j] << 16);
    *(unsigned*)&s[(nch + j) * LP + k2] = pk;
  }
}

// One K=32 step: wave computes 16(M) x 128(N): preload A frag + all 8 B frags,
// then 8 back-to-back WMMAs.
__device__ __forceinline__ void gemm_step32(const __bf16* As, const __bf16* Bs,
                                            int m0, int lrow, int sel, v8f* acc) {
  v16bf a;
  {
    const v8bf lo = *(const v8bf*)&As[(m0 + lrow) * LP + sel * 8];
    const v8bf hi = *(const v8bf*)&As[(m0 + lrow) * LP + 16 + sel * 8];
    for (int i = 0; i < 8; ++i) { a[i] = lo[i]; a[i + 8] = hi[i]; }
  }
  v16bf bm[8];
  for (int j = 0; j < 8; ++j) {
    const v8bf bl = *(const v8bf*)&Bs[(j * 16 + lrow) * LP + sel * 8];
    const v8bf bh = *(const v8bf*)&Bs[(j * 16 + lrow) * LP + 16 + sel * 8];
    for (int i = 0; i < 8; ++i) { bm[j][i] = bl[i]; bm[j][i + 8] = bh[i]; }
  }
  for (int j = 0; j < 8; ++j)
    acc[j] = __builtin_amdgcn_wmma_f32_16x16x32_bf16(
        false, a, false, bm[j], (short)0, acc[j], false, false);
}

// ---------------------------------------------------------------------------
// Kernel 1: fp32 x -> bf16 Q, per-(b,c) spatial mean into pooled[b][c],
// zero-init pooled[b][C+c].
// ---------------------------------------------------------------------------
__global__ __launch_bounds__(256) void k_convert(const float* __restrict__ x,
                                                 __bf16* __restrict__ qb,
                                                 float* __restrict__ pooled) {
  const int row = blockIdx.x;            // b*C + c
  const int b = row >> 9, c = row & 511;
  const float* xr = x + (size_t)row * HWN;
  __bf16* qr = qb + (size_t)row * HWN;
  const int t = threadIdx.x;
  float s = 0.f;
  for (int i = 0; i < 4; ++i) {
    float4 v = ((const float4*)xr)[t + i * 256];
    s += v.x + v.y + v.z + v.w;
    v4bf o;
    o[0] = (__bf16)v.x; o[1] = (__bf16)v.y; o[2] = (__bf16)v.z; o[3] = (__bf16)v.w;
    ((v4bf*)qr)[t + i * 256] = o;
  }
  for (int o = 16; o > 0; o >>= 1) s += __shfl_down(s, o, 32);
  __shared__ float red[8];
  if ((t & 31) == 0) red[t >> 5] = s;
  __syncthreads();
  if (t == 0) {
    float tot = 0.f;
    for (int i = 0; i < 8; ++i) tot += red[i];
    pooled[b * 1024 + c] = tot * (1.f / HWN);
    pooled[b * 1024 + 512 + c] = 0.f;
  }
}

// ---------------------------------------------------------------------------
// Kernel 2: energy = Q*Q^T per batch (512x512, K=4096). Upper-tri tiles only,
// mirror-store. Double-buffered TDM staging (wave 0 issues DMA), WMMA compute.
// ---------------------------------------------------------------------------
__constant__ int kTM[10] = {0,0,0,0,1,1,1,2,2,3};
__constant__ int kTN[10] = {0,1,2,3,1,2,3,2,3,3};

__global__ __launch_bounds__(256) void k_energy(const __bf16* __restrict__ qb,
                                                float* __restrict__ energy) {
  __shared__ __bf16 As[2][128 * LP];
  __shared__ __bf16 Bs[2][128 * LP];
  const int b = blockIdx.y;
  const int tileM = kTM[blockIdx.x], tileN = kTN[blockIdx.x];
  const __bf16* Q = qb + (size_t)b * Cn * HWN;
  const int t = threadIdx.x;
  const int wave = t >> 5, lane = t & 31;
  const int lrow = lane & 15, sel = lane >> 4;
  const int m0 = wave * 16;

  v8f acc[8] = {};
  const __bf16* gA = Q + (size_t)(tileM * 128) * HWN;
  const __bf16* gB = Q + (size_t)(tileN * 128) * HWN;
  constexpr int NCH = HWN / KC;   // 128 chunks

#if HAS_TDM
  const bool issuer = (wave == 0);
  if (issuer) {
    tdm_stage_rows(gA, HWN, As[0]);
    tdm_stage_rows(gB, HWN, Bs[0]);
  }
  for (int c = 0; c < NCH - 1; ++c) {
    const int cur = c & 1;
    if (issuer) __builtin_amdgcn_s_wait_tensorcnt(0);
    __syncthreads();
    if (issuer) {
      tdm_stage_rows(gA + (c + 1) * KC, HWN, As[cur ^ 1]);
      tdm_stage_rows(gB + (c + 1) * KC, HWN, Bs[cur ^ 1]);
    }
    gemm_step32(As[cur], Bs[cur], m0, lrow, sel, acc);
  }
  if (issuer) __builtin_amdgcn_s_wait_tensorcnt(0);
  __syncthreads();
  gemm_step32(As[(NCH - 1) & 1], Bs[(NCH - 1) & 1], m0, lrow, sel, acc);
#else
  stage_rows(gA, HWN, As[0], t);
  stage_rows(gB, HWN, Bs[0], t);
  for (int c = 0; c < NCH - 1; ++c) {
    const int cur = c & 1;
    wait_stage();
    __syncthreads();
    stage_rows(gA + (c + 1) * KC, HWN, As[cur ^ 1], t);
    stage_rows(gB + (c + 1) * KC, HWN, Bs[cur ^ 1], t);
    gemm_step32(As[cur], Bs[cur], m0, lrow, sel, acc);
  }
  wait_stage();
  __syncthreads();
  gemm_step32(As[(NCH - 1) & 1], Bs[(NCH - 1) & 1], m0, lrow, sel, acc);
#endif

  float* E  = energy + ((size_t)b * Cn + tileM * 128) * Cn + tileN * 128;
  float* Em = energy + ((size_t)b * Cn + tileN * 128) * Cn + tileM * 128;
  for (int j = 0; j < 8; ++j)
    for (int r = 0; r < 8; ++r) {
      const int m = m0 + r + sel * 8;
      const int n = j * 16 + lrow;
      const float v = acc[j][r];
      E[(size_t)m * Cn + n] = v;
      if (tileM != tileN) Em[(size_t)n * Cn + m] = v;
    }
}

// ---------------------------------------------------------------------------
// Kernel 3: attention = softmax(rowmax - energy) == exp(rowmin - e)/sum -> bf16
// ---------------------------------------------------------------------------
__global__ __launch_bounds__(256) void k_softmax(const float* __restrict__ energy,
                                                 __bf16* __restrict__ attnB) {
  const int row = blockIdx.x;                 // b*C + c
  const float* e = energy + (size_t)row * Cn;
  __bf16* a = attnB + (size_t)row * Cn;
  const int t = threadIdx.x;
  const float v0 = e[t], v1 = e[t + 256];
  float mn = fminf(v0, v1);
  for (int o = 16; o > 0; o >>= 1) mn = fminf(mn, __shfl_xor(mn, o, 32));
  __shared__ float redm[8];
  __shared__ float reds[8];
  if ((t & 31) == 0) redm[t >> 5] = mn;
  __syncthreads();
  float m = redm[0];
  for (int i = 1; i < 8; ++i) m = fminf(m, redm[i]);
  const float p0 = __expf(m - v0), p1 = __expf(m - v1);
  float s = p0 + p1;
  for (int o = 16; o > 0; o >>= 1) s += __shfl_xor(s, o, 32);
  if ((t & 31) == 0) reds[t >> 5] = s;
  __syncthreads();
  float tot = 0.f;
  for (int i = 0; i < 8; ++i) tot += reds[i];
  const float inv = 1.f / tot;
  a[t] = (__bf16)(p0 * inv);
  a[t + 256] = (__bf16)(p1 * inv);
}

// ---------------------------------------------------------------------------
// Kernel 4: out = attention(512x512) @ Q(512x4096). Async-to-LDS A staging,
// packed transpose B staging, double-buffered, peeled tail. Fused epilogue:
// out fp32 + pooled row-mean accumulation.
// ---------------------------------------------------------------------------
__global__ __launch_bounds__(256) void k_attnmm(const __bf16* __restrict__ attnB,
                                                const __bf16* __restrict__ qb,
                                                float* __restrict__ outb,
                                                float* __restrict__ pooled) {
  __shared__ __bf16 As[2][128 * LP];
  __shared__ __bf16 Bs[2][128 * LP];
  const int b = blockIdx.y;
  const int tileM = blockIdx.x & 3;      // c tile (4)
  const int tileN = blockIdx.x >> 2;     // n tile (32)
  const __bf16* A = attnB + (size_t)b * Cn * Cn + (size_t)(tileM * 128) * Cn;
  const __bf16* Q = qb + (size_t)b * Cn * HWN + tileN * 128;
  const int t = threadIdx.x, wave = t >> 5, lane = t & 31;
  const int lrow = lane & 15, sel = lane >> 4, m0 = wave * 16;

  v8f acc[8] = {};
  stage_rows(A, Cn, As[0], t);
  stage_transpose(Q, Bs[0], t);
  constexpr int NCH = Cn / KC;   // 16 chunks
  for (int c = 0; c < NCH - 1; ++c) {
    const int cur = c & 1;
    wait_stage();
    __syncthreads();
    stage_rows(A + (c + 1) * KC, Cn, As[cur ^ 1], t);
    stage_transpose(Q + (size_t)(c + 1) * KC * HWN, Bs[cur ^ 1], t);
    gemm_step32(As[cur], Bs[cur], m0, lrow, sel, acc);
  }
  wait_stage();
  __syncthreads();
  gemm_step32(As[(NCH - 1) & 1], Bs[(NCH - 1) & 1], m0, lrow, sel, acc);

  float* O = outb + ((size_t)b * Cn + tileM * 128) * HWN + tileN * 128;
  float rs[8];
  for (int r = 0; r < 8; ++r) rs[r] = 0.f;
  for (int j = 0; j < 8; ++j)
    for (int r = 0; r < 8; ++r) {
      const int m = m0 + r + sel * 8;
      const int n = j * 16 + lrow;
      O[(size_t)m * HWN + n] = acc[j][r];
      rs[r] += acc[j][r];
    }
  for (int o = 1; o < 16; o <<= 1)
    for (int r = 0; r < 8; ++r) rs[r] += __shfl_xor(rs[r], o, 32);
  if (lrow == 0) {
    for (int r = 0; r < 8; ++r) {
      const int m = tileM * 128 + m0 + r + sel * 8;
      atomicAdd(&pooled[b * 1024 + 512 + m], rs[r] * (1.f / HWN));
    }
  }
}

// ---------------------------------------------------------------------------
// Kernel 5: SE MLP (tiny): hidden = relu(pooled@w1^T+b1); se = sigmoid(h@w2^T+b2)
// ---------------------------------------------------------------------------
__global__ __launch_bounds__(256) void k_se(const float* __restrict__ pooled,
                                            const float* __restrict__ w1,
                                            const float* __restrict__ b1,
                                            const float* __restrict__ w2,
                                            const float* __restrict__ b2,
                                            float* __restrict__ se) {
  __shared__ float P[1024];
  __shared__ float Hd[64];
  const int b = blockIdx.x, t = threadIdx.x;
  for (int i = 0; i < 4; ++i) P[t + i * 256] = pooled[b * 1024 + t + i * 256];
  __syncthreads();
  if (t < 64) {
    float s = b1[t];
    const float* w = w1 + t * 1024;
    for (int i = 0; i < 1024; ++i) s += w[i] * P[i];
    Hd[t] = fmaxf(s, 0.f);
  }
  __syncthreads();
  for (int c = t; c < 512; c += 256) {
    float s = b2[c];
    const float* w = w2 + c * 64;
    for (int j = 0; j < 64; ++j) s += w[j] * Hd[j];
    se[b * 512 + c] = 1.f / (1.f + __expf(-s));
  }
}

// ---------------------------------------------------------------------------
// Kernel 6: y = se*x + (1-se)*out
// ---------------------------------------------------------------------------
__global__ __launch_bounds__(256) void k_blend(const float* __restrict__ x,
                                               const float* __restrict__ outb,
                                               const float* __restrict__ se,
                                               float* __restrict__ y) {
  const int row = blockIdx.x;               // b*C + c
  const float g = se[row];
  const float h = 1.f - g;
  const float4* xr = (const float4*)(x + (size_t)row * HWN);
  const float4* orr = (const float4*)(outb + (size_t)row * HWN);
  float4* yr = (float4*)(y + (size_t)row * HWN);
  const int t = threadIdx.x;
  for (int i = 0; i < 4; ++i) {
    const float4 a = xr[t + i * 256];
    const float4 o = orr[t + i * 256];
    float4 r;
    r.x = g * a.x + h * o.x;
    r.y = g * a.y + h * o.y;
    r.z = g * a.z + h * o.z;
    r.w = g * a.w + h * o.w;
    yr[t + i * 256] = r;
  }
}

extern "C" void kernel_launch(void* const* d_in, const int* in_sizes, int n_in,
                              void* d_out, int out_size, void* d_ws, size_t ws_size,
                              hipStream_t stream) {
  const float* x  = (const float*)d_in[0];
  const float* w1 = (const float*)d_in[1];
  const float* b1 = (const float*)d_in[2];
  const float* w2 = (const float*)d_in[3];
  const float* b2 = (const float*)d_in[4];
  float* y = (float*)d_out;

  char* ws = (char*)d_ws;
  __bf16* qb     = (__bf16*)(ws);                      //  64 MiB  bf16 Q
  __bf16* attnB  = (__bf16*)(ws + 67108864);           //   8 MiB  bf16 attention
  float*  energy = (float*)(ws + 75497472);            //  16 MiB  f32 energy
  float*  outb   = (float*)(ws + 92274688);            // 128 MiB  f32 out
  float*  pooled = (float*)(ws + 226492416);           //  64 KiB  [B][2C]
  float*  seb    = (float*)(ws + 226557952);           //  32 KiB  [B][C]

  k_convert<<<Bn * Cn, 256, 0, stream>>>(x, qb, pooled);
  k_energy <<<dim3(10, Bn), 256, 0, stream>>>(qb, energy);
  k_softmax<<<Bn * Cn, 256, 0, stream>>>(energy, attnB);
  k_attnmm <<<dim3(128, Bn), 256, 0, stream>>>(attnB, qb, outb, pooled);
  k_se     <<<Bn, 256, 0, stream>>>(pooled, w1, b1, w2, b2, seb);
  k_blend  <<<Bn * Cn, 256, 0, stream>>>(x, outb, seb, y);
}
